// GroupedQueryAttention_66494683677314
// MI455X (gfx1250) — matrix operand order repaired and checked by reading, not compiled, not run on hardware
//
#include <hip/hip_runtime.h>

typedef __attribute__((ext_vector_type(16))) __bf16 v16bf;
typedef __attribute__((ext_vector_type(8)))  float  v8f;

union Frag16 {
    v16bf v;
    uint4 q[2];
};

static __device__ __forceinline__ float bf2f(unsigned short u) {
    union { unsigned int i; float f; } c;
    c.i = ((unsigned int)u) << 16;
    return c.f;
}
static __device__ __forceinline__ unsigned short f2bf(float f) {
    union { float f; unsigned int i; } c;
    c.f = f;
    unsigned int u = c.i;
    u += 0x7FFFu + ((u >> 16) & 1u);   // round-to-nearest-even
    return (unsigned short)(u >> 16);
}

// ---------------------------------------------------------------------------
// f32 -> bf16 cast, 2 elements per thread (packed dword stores)
// ---------------------------------------------------------------------------
__global__ void cast_f32_bf16_kernel(const float* __restrict__ in,
                                     unsigned int* __restrict__ out, int npairs) {
    int i = blockIdx.x * blockDim.x + threadIdx.x;
    if (i >= npairs) return;
    float2 f = reinterpret_cast<const float2*>(in)[i];
    unsigned int lo = f2bf(f.x), hi = f2bf(f.y);
    out[i] = lo | (hi << 16);
}

// ---------------------------------------------------------------------------
// C[M,N] = A[M,K] @ B[N,K]^T   (bf16 in, f32 accumulate, bf16 or f32 out)
// 128x128 tile per block, 8 waves, each wave owns a 16-row stripe.
// B tile (128x32 bf16 = 8KB) staged in LDS and shared by all waves.
// ---------------------------------------------------------------------------
template <bool OUT_BF16>
__global__ __launch_bounds__(256)
void gemm_bf16_kernel(const unsigned short* __restrict__ A,
                      const unsigned short* __restrict__ Bm,
                      void* __restrict__ Cout,
                      int M, int N, int K) {
    __shared__ __align__(32) unsigned short ldsB[128 * 32];

    const int tid  = threadIdx.x;
    const int wave = tid >> 5;
    const int lane = tid & 31;
    const int cl   = lane & 15;      // column lane / row-in-frag
    const int hi   = lane >> 4;      // half-wave select
    const int kb0  = hi ? 8 : 0;     // A-frag K base (16-bit A layout)

    const int row0 = blockIdx.y * 128;
    const int col0 = blockIdx.x * 128;
    const size_t arow = (size_t)(row0 + wave * 16 + cl);

    v8f acc[8];
    for (int i = 0; i < 8; ++i)
        for (int j = 0; j < 8; ++j) acc[i][j] = 0.f;

    for (int k0 = 0; k0 < K; k0 += 32) {
        __syncthreads();
        // cooperatively stage B tile: rows col0..col0+127, cols k0..k0+31
#pragma unroll
        for (int i = 0; i < 2; ++i) {
            int cidx = tid + i * 256;            // 512 chunks of 8 ushorts
            int n = cidx >> 2;
            int c = cidx & 3;
            const uint4* src = reinterpret_cast<const uint4*>(
                &Bm[(size_t)(col0 + n) * K + k0 + c * 8]);
            *reinterpret_cast<uint4*>(&ldsB[n * 32 + c * 8]) = *src;
        }
        __syncthreads();

        // A fragment (16x32 bf16, documented wave32 layout)
        Frag16 a;
        const unsigned short* ap = &A[arow * K + k0 + kb0];
        a.q[0] = *reinterpret_cast<const uint4*>(ap);
        a.q[1] = *reinterpret_cast<const uint4*>(ap + 16);
        if (k0 + 32 < K) __builtin_prefetch(ap + 32, 0, 1);

#pragma unroll
        for (int nt = 0; nt < 8; ++nt) {
            Frag16 b;
            const unsigned short* bp = &ldsB[(nt * 16 + cl) * 32 + (hi ? 8 : 0)];
            b.q[0] = *reinterpret_cast<const uint4*>(bp);
            b.q[1] = *reinterpret_cast<const uint4*>(bp + 16);
            acc[nt] = __builtin_amdgcn_wmma_f32_16x16x32_bf16(
                false, a.v, false, b.v, (short)0, acc[nt], false, false);
        }
    }

    const int rb = hi ? 8 : 0;
#pragma unroll
    for (int nt = 0; nt < 8; ++nt) {
        const int gcol = col0 + nt * 16 + cl;
#pragma unroll
        for (int r = 0; r < 8; ++r) {
            const size_t grow = (size_t)(row0 + wave * 16 + rb + r);
            if (OUT_BF16)
                reinterpret_cast<unsigned short*>(Cout)[grow * N + gcol] = f2bf(acc[nt][r]);
            else
                reinterpret_cast<float*>(Cout)[grow * N + gcol] = acc[nt][r];
        }
    }
}

// ---------------------------------------------------------------------------
// Interleaved RoPE, in place on bf16 [B,T,nh,128]
// ---------------------------------------------------------------------------
__global__ void rope_kernel(unsigned short* __restrict__ data,
                            const float* __restrict__ freqs,
                            int Bv, int T, int nh) {
    int idx = blockIdx.x * blockDim.x + threadIdx.x;
    int total = Bv * T * nh * 64;
    if (idx >= total) return;
    int i = idx & 63;
    int h = (idx >> 6) % nh;
    int t = (idx / (64 * nh)) % T;
    int b = idx / (64 * nh * T);
    float f = freqs[t * 64 + i];
    float c = __cosf(f), s = __sinf(f);
    size_t base = ((size_t)(b * T + t) * nh + h) * 128 + 2 * i;
    float x1 = bf2f(data[base]);
    float x2 = bf2f(data[base + 1]);
    data[base]     = f2bf(x1 * c - x2 * s);
    data[base + 1] = f2bf(x1 * s + x2 * c);
}

// ---------------------------------------------------------------------------
// bf16 transpose: in [Bv][T][Cn]  ->  out [Bv][Cn][T]
// 32x32 tiles through padded LDS; coalesced 8B vector IO on both sides.
// ---------------------------------------------------------------------------
__global__ __launch_bounds__(256)
void transpose_bf16_kernel(const unsigned short* __restrict__ in,
                           unsigned short* __restrict__ out,
                           int T, int Cn) {
    __shared__ unsigned short tile[32][33];

    const int tid = threadIdx.x;
    const int t0 = blockIdx.x * 32;
    const int c0 = blockIdx.y * 32;
    const int b  = blockIdx.z;

    {
        int tr  = tid >> 3;          // 0..31 rows (t)
        int tc4 = (tid & 7) * 4;     // 4 cols per thread
        union { unsigned short s[4]; uint2 u; } v;
        v.u = *reinterpret_cast<const uint2*>(
            &in[((size_t)b * T + t0 + tr) * Cn + c0 + tc4]);
#pragma unroll
        for (int j = 0; j < 4; ++j) tile[tr][tc4 + j] = v.s[j];
    }
    __syncthreads();
    {
        int oc  = tid >> 3;          // 0..31 rows of output (c)
        int ot4 = (tid & 7) * 4;     // 4 t per thread
        union { unsigned short s[4]; uint2 u; } v;
#pragma unroll
        for (int j = 0; j < 4; ++j) v.s[j] = tile[ot4 + j][oc];
        *reinterpret_cast<uint2*>(
            &out[((size_t)b * Cn + c0 + oc) * T + t0 + ot4]) = v.u;
    }
}

// ---------------------------------------------------------------------------
// Flash attention: one block = 128 query rows of one (b,h).
// 8 waves x 16 query rows. Key blocks of 32, causal, online softmax.
// QK^T and P@V both on v_wmma_f32_16x16x32_bf16.
// V is consumed pre-transposed: Vt[b][kv_channel][t]  (t-contiguous rows),
// so both K and V^T tiles stage into LDS with b128 vector copies.
// ---------------------------------------------------------------------------
__global__ __launch_bounds__(256)
void flash_attn_kernel(const unsigned short* __restrict__ Q,
                       const unsigned short* __restrict__ Kc,
                       const unsigned short* __restrict__ Vt,
                       unsigned short* __restrict__ O,
                       int T) {
    constexpr int NH = 16, NKV = 4, DH = 128;
    constexpr float scale = 0.08838834764831845f;  // 1/sqrt(128)
    constexpr float NEG = -3.0e38f;

    __shared__ __align__(32) unsigned short ldsK[32 * 128];   // [s][d]
    __shared__ __align__(32) unsigned short ldsV[128 * 32];   // [d][s]
    __shared__ __align__(32) unsigned short ldsP[8 * 16 * 32];// per-wave P tile [m][s]

    const int tid = threadIdx.x;
    const int wave = tid >> 5, lane = tid & 31;
    const int cl = lane & 15, hi = lane >> 4;
    const int rb  = hi ? 8 : 0;
    const int kb0 = hi ? 8 : 0;

    const int q0 = blockIdx.x * 128;
    const int bh = blockIdx.y;
    const int b = bh / NH, h = bh % NH, kvh = h / (NH / NKV);
    const size_t rowbase = (size_t)b * T;

    // Q fragments: 4 K-chunks of 32 over D=128, loaded once
    Frag16 qa[4];
    {
        const int arow = q0 + wave * 16 + cl;
        const unsigned short* qp = &Q[(rowbase + arow) * (NH * DH) + h * DH];
#pragma unroll
        for (int c4 = 0; c4 < 4; ++c4) {
            qa[c4].q[0] = *reinterpret_cast<const uint4*>(qp + c4 * 32 + kb0);
            qa[c4].q[1] = *reinterpret_cast<const uint4*>(qp + c4 * 32 + kb0 + 16);
        }
    }

    v8f o[8];
    for (int i = 0; i < 8; ++i)
        for (int j = 0; j < 8; ++j) o[i][j] = 0.f;
    float mrow[8], lrow[8];
    for (int r = 0; r < 8; ++r) { mrow[r] = NEG; lrow[r] = 0.f; }

    const int s_end = q0 + 128;
    for (int sb = 0; sb < s_end; sb += 32) {
        __syncthreads();
        // stage K block [32 x 128], rows d-contiguous
#pragma unroll
        for (int i = 0; i < 2; ++i) {
            int cidx = tid + i * 256;
            int s = cidx >> 4, c = cidx & 15;
            const uint4* src = reinterpret_cast<const uint4*>(
                &Kc[(rowbase + sb + s) * (NKV * DH) + kvh * DH + c * 8]);
            *reinterpret_cast<uint4*>(&ldsK[s * 128 + c * 8]) = *src;
        }
        // stage V^T block [128 x 32] from pre-transposed Vt, rows s-contiguous
#pragma unroll
        for (int i = 0; i < 2; ++i) {
            int cidx = tid + i * 256;            // 512 chunks of 8 ushorts
            int d = cidx >> 2, c = cidx & 3;
            const uint4* src = reinterpret_cast<const uint4*>(
                &Vt[((size_t)b * (NKV * DH) + kvh * DH + d) * T + sb + c * 8]);
            *reinterpret_cast<uint4*>(&ldsV[d * 32 + c * 8]) = *src;
        }
        __syncthreads();

        // S = Q K^T : two 16-key column tiles, K-dim = 128 in 4 wmma steps
        v8f sc[2];
        for (int i = 0; i < 2; ++i)
            for (int j = 0; j < 8; ++j) sc[i][j] = 0.f;
#pragma unroll
        for (int c4 = 0; c4 < 4; ++c4) {
#pragma unroll
            for (int nt = 0; nt < 2; ++nt) {
                Frag16 bk;
                const unsigned short* bp =
                    &ldsK[(nt * 16 + cl) * 128 + c4 * 32 + (hi ? 8 : 0)];
                bk.q[0] = *reinterpret_cast<const uint4*>(bp);
                bk.q[1] = *reinterpret_cast<const uint4*>(bp + 16);
                sc[nt] = __builtin_amdgcn_wmma_f32_16x16x32_bf16(
                    false, qa[c4].v, false, bk.v, (short)0, sc[nt], false, false);
            }
        }

        // scale + causal mask + per-row tile max
        const int qg0 = q0 + wave * 16 + rb;
        float tmax[8];
#pragma unroll
        for (int r = 0; r < 8; ++r) {
            float mx = NEG;
#pragma unroll
            for (int nt = 0; nt < 2; ++nt) {
                int scol = sb + nt * 16 + cl;
                float v = sc[nt][r] * scale;
                if (scol > qg0 + r) v = NEG;
                sc[nt][r] = v;
                mx = fmaxf(mx, v);
            }
            tmax[r] = mx;
        }
#pragma unroll
        for (int off = 1; off < 16; off <<= 1) {
#pragma unroll
            for (int r = 0; r < 8; ++r)
                tmax[r] = fmaxf(tmax[r], __shfl_xor(tmax[r], off, 32));
        }

        // online softmax update; emit P (bf16) into this wave's LDS tile
        float alpha[8], psum[8];
#pragma unroll
        for (int r = 0; r < 8; ++r) {
            float mn = fmaxf(mrow[r], tmax[r]);
            alpha[r] = __expf(mrow[r] - mn);
            mrow[r] = mn;
            float ps = 0.f;
#pragma unroll
            for (int nt = 0; nt < 2; ++nt) {
                float p = __expf(sc[nt][r] - mn);
                ps += p;
                ldsP[wave * 512 + (rb + r) * 32 + nt * 16 + cl] = f2bf(p);
            }
            psum[r] = ps;
        }
#pragma unroll
        for (int off = 1; off < 16; off <<= 1) {
#pragma unroll
            for (int r = 0; r < 8; ++r)
                psum[r] += __shfl_xor(psum[r], off, 32);
        }
#pragma unroll
        for (int r = 0; r < 8; ++r) lrow[r] = lrow[r] * alpha[r] + psum[r];

        // rescale running output
#pragma unroll
        for (int j = 0; j < 8; ++j) {
#pragma unroll
            for (int r = 0; r < 8; ++r) o[j][r] *= alpha[r];
        }

        // our own wave reads its P tile back: wait for the DS stores
        asm volatile("s_wait_dscnt 0" ::: "memory");

        // O += P V : P is A (16x32), V^T tiles are B, 8 D-tiles of 16
        Frag16 pa;
        const unsigned short* pp = &ldsP[wave * 512 + cl * 32 + kb0];
        pa.q[0] = *reinterpret_cast<const uint4*>(pp);
        pa.q[1] = *reinterpret_cast<const uint4*>(pp + 16);
#pragma unroll
        for (int j = 0; j < 8; ++j) {
            Frag16 bv;
            const unsigned short* bp = &ldsV[(j * 16 + cl) * 32 + (hi ? 8 : 0)];
            bv.q[0] = *reinterpret_cast<const uint4*>(bp);
            bv.q[1] = *reinterpret_cast<const uint4*>(bp + 16);
            o[j] = __builtin_amdgcn_wmma_f32_16x16x32_bf16(
                false, pa.v, false, bv.v, (short)0, o[j], false, false);
        }
    }

    // normalize and store bf16 [b*T+t, h*128+d]
#pragma unroll
    for (int j = 0; j < 8; ++j) {
#pragma unroll
        for (int r = 0; r < 8; ++r) {
            float inv = 1.0f / lrow[r];
            size_t row = rowbase + q0 + wave * 16 + rb + r;
            O[row * (NH * DH) + h * DH + j * 16 + cl] = f2bf(o[j][r] * inv);
        }
    }
}

// ---------------------------------------------------------------------------
// Host-side orchestration
// ---------------------------------------------------------------------------
extern "C" void kernel_launch(void* const* d_in, const int* in_sizes, int n_in,
                              void* d_out, int out_size, void* d_ws, size_t ws_size,
                              hipStream_t stream) {
    (void)in_sizes; (void)n_in; (void)out_size; (void)ws_size;
    const float* x  = (const float*)d_in[0];
    const float* fr = (const float*)d_in[1];
    const float* wq = (const float*)d_in[2];
    const float* wk = (const float*)d_in[3];
    const float* wv = (const float*)d_in[4];
    const float* wo = (const float*)d_in[5];

    const int Bv = 2, T = 2048, C = 2048;
    const int NH = 16, NKV = 4, DH = 128;
    const int M    = Bv * T;       // 4096
    const int NQ   = NH * DH;      // 2048
    const int NKVd = NKV * DH;     // 512

    unsigned short* p = (unsigned short*)d_ws;
    unsigned short* xb  = p; p += (size_t)M * C;
    unsigned short* wqb = p; p += (size_t)NQ * C;
    unsigned short* wkb = p; p += (size_t)NKVd * C;
    unsigned short* wvb = p; p += (size_t)NKVd * C;
    unsigned short* wob = p; p += (size_t)C * C;
    unsigned short* qb  = p; p += (size_t)M * NQ;
    unsigned short* kb  = p; p += (size_t)M * NKVd;
    unsigned short* vb  = p; p += (size_t)M * NKVd;
    unsigned short* vtb = p; p += (size_t)M * NKVd;   // V^T: [b][c][t]
    unsigned short* ob  = p; p += (size_t)M * C;

    auto cast = [&](const float* src, unsigned short* dst, size_t n) {
        size_t np = n / 2;
        cast_f32_bf16_kernel<<<dim3((unsigned)((np + 255) / 256)), dim3(256), 0, stream>>>(
            src, (unsigned int*)dst, (int)np);
    };
    cast(x,  xb,  (size_t)M * C);
    cast(wq, wqb, (size_t)NQ * C);
    cast(wk, wkb, (size_t)NKVd * C);
    cast(wv, wvb, (size_t)NKVd * C);
    cast(wo, wob, (size_t)C * C);

    gemm_bf16_kernel<true><<<dim3(NQ / 128,   M / 128), 256, 0, stream>>>(xb, wqb, qb, M, NQ,   C);
    gemm_bf16_kernel<true><<<dim3(NKVd / 128, M / 128), 256, 0, stream>>>(xb, wkb, kb, M, NKVd, C);
    gemm_bf16_kernel<true><<<dim3(NKVd / 128, M / 128), 256, 0, stream>>>(xb, wvb, vb, M, NKVd, C);

    {
        int tot = Bv * T * NH * 64;
        rope_kernel<<<(tot + 255) / 256, 256, 0, stream>>>(qb, fr, Bv, T, NH);
        tot = Bv * T * NKV * 64;
        rope_kernel<<<(tot + 255) / 256, 256, 0, stream>>>(kb, fr, Bv, T, NKV);
    }

    // one-time V transpose so attention stages V^T with b128 vector copies
    transpose_bf16_kernel<<<dim3(T / 32, NKVd / 32, Bv), 256, 0, stream>>>(vb, vtb, T, NKVd);

    flash_attn_kernel<<<dim3(T / 128, Bv * NH), 256, 0, stream>>>(qb, kb, vtb, ob, T);

    gemm_bf16_kernel<false><<<dim3(C / 128, M / 128), 256, 0, stream>>>(ob, wob, d_out, M, C, C);
}
